// GCNEncoder_12386685681960
// MI455X (gfx1250) — compile-verified, matrix operand order checked by reading, not run on hardware
//
#include <hip/hip_runtime.h>
#include <hip/hip_bf16.h>
#include <math.h>

// ---------------------------------------------------------------------------
// GCN encoder for MI455X (gfx1250, wave32).
//   layer: H = X@W ; AGG = D^-1/2 (A+I) D^-1/2 H ; LN ; (GELU)
// GEMMs via v_wmma_f32_16x16x32_bf16 (bf16 in, f32 acc).
// Edge aggregation via global_atomic_add_f32 (unsafeAtomicAdd), 1 wave/edge.
// ---------------------------------------------------------------------------

typedef __attribute__((ext_vector_type(16))) __bf16 v16bf;
typedef __attribute__((ext_vector_type(8)))  float  v8f;

#define IN_DIM 256
#define KDIM   256   // both GEMMs have K = 256

static inline unsigned cdivu(unsigned a, unsigned b) { return (a + b - 1) / b; }

// ---- degree / normalization ------------------------------------------------
__global__ __launch_bounds__(256) void k_deg_init(float* deg, int n) {
  int i = blockIdx.x * blockDim.x + threadIdx.x;
  if (i < n) deg[i] = 1.0f;            // self-loop contributes 1 to in-degree
}

__global__ __launch_bounds__(256) void k_deg_edges(float* deg, const int* dst, int e) {
  int i = blockIdx.x * blockDim.x + threadIdx.x;
  if (i < e) unsafeAtomicAdd(&deg[dst[i]], 1.0f);
}

__global__ __launch_bounds__(256) void k_dinv(float* deg, int n) {
  int i = blockIdx.x * blockDim.x + threadIdx.x;
  if (i < n) { float d = deg[i]; deg[i] = d > 0.0f ? rsqrtf(d) : 0.0f; }
}

// ---- fp32 -> bf16 conversion (activations) --------------------------------
__global__ __launch_bounds__(256) void k_cvt_bf16(const float* __restrict__ x,
                                                  __hip_bfloat16* __restrict__ o,
                                                  long long n4) {
  long long t = (long long)blockIdx.x * blockDim.x + threadIdx.x;
  if (t >= n4) return;
  float4 v = *(const float4*)(x + t * 4);
  union { __hip_bfloat16 h[4]; uint2 u; } p;
  p.h[0] = __float2bfloat16(v.x); p.h[1] = __float2bfloat16(v.y);
  p.h[2] = __float2bfloat16(v.z); p.h[3] = __float2bfloat16(v.w);
  *(uint2*)(o + t * 4) = p.u;
}

// ---- W [K,OUT] (f32, row-major) -> Wt [OUT,K] (bf16) ----------------------
__global__ __launch_bounds__(256) void k_wt(const float* __restrict__ W,
                                            __hip_bfloat16* __restrict__ Wt,
                                            int out_dim) {
  int t = blockIdx.x * blockDim.x + threadIdx.x;  // t = n*K + k
  if (t >= out_dim * KDIM) return;
  int k = t & (KDIM - 1);
  int n = t >> 8;
  Wt[t] = __float2bfloat16(W[k * out_dim + n]);
}

// ---- WMMA GEMM:  C[M,OUT] = A[M,256](bf16) * Wt[OUT,256](bf16)^T ----------
// One wave computes one 16x16 output tile; 8 wmma steps over K=256.
__global__ __launch_bounds__(256) void k_gemm_bf16(const __bf16* __restrict__ A,
                                                   const __bf16* __restrict__ Bt,
                                                   float* __restrict__ C,
                                                   int mtiles, int ntiles) {
  int wave = blockIdx.x * (blockDim.x >> 5) + (threadIdx.x >> 5);
  if (wave >= mtiles * ntiles) return;
  int mt = wave / ntiles;
  int nt = wave - mt * ntiles;
  int lane  = threadIdx.x & 31;
  int mn    = lane & 15;                // row of A / col of B this lane carries
  int kbase = (lane >> 4) << 3;         // 0 for lanes 0-15, 8 for lanes 16-31

  const __bf16* arow = A  + (size_t)(mt * 16 + mn) * KDIM + kbase;
  const __bf16* brow = Bt + (size_t)(nt * 16 + mn) * KDIM + kbase;

  v8f acc = {};
#pragma unroll
  for (int k0 = 0; k0 < KDIM; k0 += 32) {
    // ISA 16-bit A layout: lane holds K=[kbase,kbase+8) then K=[kbase+16,kbase+24)
    union { uint4 u[2]; v16bf v; } fa, fb;
    fa.u[0] = *(const uint4*)(arow + k0);
    fa.u[1] = *(const uint4*)(arow + k0 + 16);
    fb.u[0] = *(const uint4*)(brow + k0);
    fb.u[1] = *(const uint4*)(brow + k0 + 16);
    acc = __builtin_amdgcn_wmma_f32_16x16x32_bf16(
        /*neg_a=*/false, fa.v, /*neg_b=*/false, fb.v,
        /*c_mod=*/(short)0, acc, /*reuse_a=*/false, /*reuse_b=*/false);
  }

  // C/D layout: c[r] -> row = r + 8*(lane>>4), col = lane&15
  int out_dim = ntiles << 4;
  int rbase = mt * 16 + ((lane >> 4) << 3);
  int col   = nt * 16 + (lane & 15);
#pragma unroll
  for (int r = 0; r < 8; ++r)
    C[(size_t)(rbase + r) * out_dim + col] = acc[r];
}

// ---- self-loop init:  AGG[i,f] = H[i,f] * dinv[i]^2 -----------------------
__global__ __launch_bounds__(256) void k_selfloop(const float* __restrict__ H,
                                                  float* __restrict__ AGG,
                                                  const float* __restrict__ dinv,
                                                  int F, long long total4) {
  long long t = (long long)blockIdx.x * blockDim.x + threadIdx.x;
  if (t >= total4) return;
  long long idx = t * 4;
  int row = (int)(idx / F);
  float s = dinv[row]; s *= s;
  float4 v = *(const float4*)(H + idx);
  v.x *= s; v.y *= s; v.z *= s; v.w *= s;
  *(float4*)(AGG + idx) = v;
}

// ---- edge scatter: AGG[dst] += H[src] * dinv[src]*dinv[dst] ---------------
// One wave32 per edge; F/32 floats per lane; f32 hardware atomics.
__global__ __launch_bounds__(256) void k_scatter(const float* __restrict__ H,
                                                 float* __restrict__ AGG,
                                                 const int* __restrict__ src,
                                                 const int* __restrict__ dst,
                                                 const float* __restrict__ dinv,
                                                 int e, int F) {
  int w = blockIdx.x * (blockDim.x >> 5) + (threadIdx.x >> 5);
  if (w >= e) return;
  int lane = threadIdx.x & 31;
  int s = src[w], d = dst[w];
  float norm = dinv[s] * dinv[d];
  int per = F >> 5;                                   // 8 (F=256) or 4 (F=128)
  const float* hp = H   + (size_t)s * F + lane * per;
  float*       op = AGG + (size_t)d * F + lane * per;
#pragma unroll 2
  for (int i = 0; i < per; i += 4) {
    float4 v = *(const float4*)(hp + i);
    unsafeAtomicAdd(op + i + 0, v.x * norm);
    unsafeAtomicAdd(op + i + 1, v.y * norm);
    unsafeAtomicAdd(op + i + 2, v.z * norm);
    unsafeAtomicAdd(op + i + 3, v.w * norm);
  }
}

// ---- wave32 tree reduction -------------------------------------------------
__device__ inline float wave_sum32(float v) {
#pragma unroll
  for (int m = 16; m >= 1; m >>= 1) v += __shfl_xor(v, m, 32);
  return v;
}

// ---- layer1 epilogue: bias + LayerNorm + exact GELU -> bf16 ---------------
__global__ __launch_bounds__(256) void k_ln_gelu_bf16(const float* __restrict__ AGG,
                                                      const float* __restrict__ b,
                                                      const float* __restrict__ g,
                                                      const float* __restrict__ be,
                                                      __hip_bfloat16* __restrict__ X2,
                                                      int n) {
  int row = blockIdx.x * (blockDim.x >> 5) + (threadIdx.x >> 5);
  if (row >= n) return;
  int lane = threadIdx.x & 31;
  const float* a = AGG + (size_t)row * 256 + lane * 8;
  float v[8];
  float4 p0 = *(const float4*)(a);
  float4 p1 = *(const float4*)(a + 4);
  float4 q0 = *(const float4*)(b + lane * 8);
  float4 q1 = *(const float4*)(b + lane * 8 + 4);
  v[0] = p0.x + q0.x; v[1] = p0.y + q0.y; v[2] = p0.z + q0.z; v[3] = p0.w + q0.w;
  v[4] = p1.x + q1.x; v[5] = p1.y + q1.y; v[6] = p1.z + q1.z; v[7] = p1.w + q1.w;
  float s = 0.f, ss = 0.f;
#pragma unroll
  for (int i = 0; i < 8; ++i) { s += v[i]; ss += v[i] * v[i]; }
  s  = wave_sum32(s);
  ss = wave_sum32(ss);
  float mean = s * (1.0f / 256.0f);
  float var  = ss * (1.0f / 256.0f) - mean * mean;
  float rstd = rsqrtf(var + 1e-5f);
  union { __hip_bfloat16 h[8]; uint4 u; } o;
#pragma unroll
  for (int i = 0; i < 8; ++i) {
    float y = (v[i] - mean) * rstd * g[lane * 8 + i] + be[lane * 8 + i];
    y = 0.5f * y * (1.0f + erff(y * 0.70710678118654752f));   // exact GELU
    o.h[i] = __float2bfloat16(y);
  }
  *(uint4*)(X2 + (size_t)row * 256 + lane * 8) = o.u;
}

// ---- layer2 epilogue: bias + LayerNorm (in-place on d_out, F=128) ---------
__global__ __launch_bounds__(256) void k_ln_final(float* __restrict__ OUT,
                                                  const float* __restrict__ b,
                                                  const float* __restrict__ g,
                                                  const float* __restrict__ be,
                                                  int n) {
  int row = blockIdx.x * (blockDim.x >> 5) + (threadIdx.x >> 5);
  if (row >= n) return;
  int lane = threadIdx.x & 31;
  float* a = OUT + (size_t)row * 128 + lane * 4;
  float4 p = *(const float4*)(a);
  float4 q = *(const float4*)(b + lane * 4);
  float v[4] = { p.x + q.x, p.y + q.y, p.z + q.z, p.w + q.w };
  float s = 0.f, ss = 0.f;
#pragma unroll
  for (int i = 0; i < 4; ++i) { s += v[i]; ss += v[i] * v[i]; }
  s  = wave_sum32(s);
  ss = wave_sum32(ss);
  float mean = s * (1.0f / 128.0f);
  float var  = ss * (1.0f / 128.0f) - mean * mean;
  float rstd = rsqrtf(var + 1e-5f);
  float4 o;
  o.x = (v[0] - mean) * rstd * g[lane * 4 + 0] + be[lane * 4 + 0];
  o.y = (v[1] - mean) * rstd * g[lane * 4 + 1] + be[lane * 4 + 1];
  o.z = (v[2] - mean) * rstd * g[lane * 4 + 2] + be[lane * 4 + 2];
  o.w = (v[3] - mean) * rstd * g[lane * 4 + 3] + be[lane * 4 + 3];
  *(float4*)(a) = o;
}

// ---------------------------------------------------------------------------
extern "C" void kernel_launch(void* const* d_in, const int* in_sizes, int n_in,
                              void* d_out, int out_size, void* d_ws, size_t ws_size,
                              hipStream_t stream) {
  const float* x   = (const float*)d_in[0];
  const int*   ei  = (const int*)  d_in[1];
  const float* W1  = (const float*)d_in[2];
  const float* b1  = (const float*)d_in[3];
  const float* g1  = (const float*)d_in[4];
  const float* be1 = (const float*)d_in[5];
  const float* W2  = (const float*)d_in[6];
  const float* b2  = (const float*)d_in[7];
  const float* g2  = (const float*)d_in[8];
  const float* be2 = (const float*)d_in[9];
  float* out = (float*)d_out;

  const int N = in_sizes[0] / IN_DIM;     // 100000
  const int E = in_sizes[1] / 2;          // 3200000
  const int* src = ei;
  const int* dst = ei + E;

  // workspace layout (bytes), 256B-aligned slices
  char* p = (char*)d_ws;
  auto take = [&](size_t bytes) { char* r = p; p += (bytes + 255) & ~(size_t)255; return r; };
  float*          dinv = (float*)         take((size_t)N * 4);
  __hip_bfloat16* Xb   = (__hip_bfloat16*)take((size_t)N * 256 * 2); // X bf16; reused as X2 bf16
  __hip_bfloat16* W1t  = (__hip_bfloat16*)take((size_t)256 * 256 * 2);
  __hip_bfloat16* W2t  = (__hip_bfloat16*)take((size_t)128 * 256 * 2);
  float*          H    = (float*)         take((size_t)N * 256 * 4); // H1; reused as H2
  float*          AGG1 = (float*)         take((size_t)N * 256 * 4);
  // AGG2 lives in d_out

  const int B = 256;

  // --- normalization coefficients ---
  k_deg_init <<<cdivu(N, B), B, 0, stream>>>(dinv, N);
  k_deg_edges<<<cdivu(E, B), B, 0, stream>>>(dinv, dst, E);
  k_dinv     <<<cdivu(N, B), B, 0, stream>>>(dinv, N);

  // --- bf16 conversions ---
  long long x4 = (long long)N * 256 / 4;
  k_cvt_bf16<<<cdivu((unsigned)x4, B), B, 0, stream>>>(x, Xb, x4);
  k_wt<<<cdivu(256 * 256, B), B, 0, stream>>>(W1, W1t, 256);
  k_wt<<<cdivu(128 * 256, B), B, 0, stream>>>(W2, W2t, 128);

  // --- layer 1 ---
  {
    int mt = cdivu(N, 16), nt = 256 / 16;                     // 6250 x 16 tiles
    k_gemm_bf16<<<cdivu(mt * nt, 8), B, 0, stream>>>((const __bf16*)Xb, (const __bf16*)W1t, H, mt, nt);
    long long t4 = (long long)N * 256 / 4;
    k_selfloop<<<cdivu((unsigned)t4, B), B, 0, stream>>>(H, AGG1, dinv, 256, t4);
    k_scatter<<<cdivu(E, 8), B, 0, stream>>>(H, AGG1, src, dst, dinv, E, 256);
    k_ln_gelu_bf16<<<cdivu(N, 8), B, 0, stream>>>(AGG1, b1, g1, be1, Xb, N); // Xb now holds X2(bf16)
  }

  // --- layer 2 ---
  {
    int mt = cdivu(N, 16), nt = 128 / 16;                     // 6250 x 8 tiles
    k_gemm_bf16<<<cdivu(mt * nt, 8), B, 0, stream>>>((const __bf16*)Xb, (const __bf16*)W2t, H, mt, nt);
    long long t4 = (long long)N * 128 / 4;
    k_selfloop<<<cdivu((unsigned)t4, B), B, 0, stream>>>(H, out, dinv, 128, t4);
    k_scatter<<<cdivu(E, 8), B, 0, stream>>>(H, out, src, dst, dinv, E, 128);
    k_ln_final<<<cdivu(N, 8), B, 0, stream>>>(out, b2, g2, be2, N);
  }
}
